// Transformer_LM_20306605375520
// MI455X (gfx1250) — matrix operand order repaired and checked by reading, not compiled, not run on hardware
//
#include <hip/hip_runtime.h>
#include <hip/hip_bf16.h>
#include <math.h>

typedef __bf16 bf16;
typedef __bf16 v16bf __attribute__((ext_vector_type(16)));
typedef float  v8f   __attribute__((ext_vector_type(8)));
typedef int    v4i   __attribute__((ext_vector_type(4)));

#define D_MODEL 1024
#define N_LAYERS 8
#define N_HEADS 16
#define D_HEAD 64
#define D_FF 4096
#define BATCH 2
#define SEQ 1024
#define VOCAB 32000
#define RMS_EPS 1e-5f

// GEMM tiling (generic weight GEMMs): block 128x128, wave 32x64, BK=32
#define BM 128
#define BNG 128
// Attention tiling: block 128x64 (Dk=64)
#define BNA 64
#define BK 32
#define LDP 40   // LDS pitch in bf16 (32 + 8 pad) -> 80B rows, 16B aligned

enum { EPI_F32 = 0, EPI_F32_RES = 1, EPI_BF16 = 2, EPI_BF16_GELU = 3 };

union FragU { uint4 u[2]; v16bf v; };
union AccU  { v8f v; float f[8]; };
union Pack8 { uint4 u; bf16 h[8]; };

// ---------------- async global->LDS (CDNA5 ASYNCcnt path), with fallback ----------------
#if defined(__gfx1250__) && __has_builtin(__builtin_amdgcn_global_load_async_to_lds_b128)
#define HAVE_ASYNC_LDS 1
__device__ __forceinline__ void async_cp16(const bf16* g, bf16* l) {
    __builtin_amdgcn_global_load_async_to_lds_b128(
        (__attribute__((address_space(1))) v4i*)(bf16*)g,
        (__attribute__((address_space(3))) v4i*)l,
        /*offset=*/0, /*cpol=*/0);
}
__device__ __forceinline__ void async_wait0() {
#if __has_builtin(__builtin_amdgcn_s_wait_asynccnt)
    __builtin_amdgcn_s_wait_asynccnt(0);
#else
    asm volatile("s_wait_asynccnt 0" ::: "memory");
#endif
}
#else
#define HAVE_ASYNC_LDS 0
#endif

// ---- WMMA fragment loads from LDS (layouts per CDNA5 ISA 7.12.2) ----
// A (16x32 bf16): lane m = lane%16; half=lane>>4; k = (e%8) + 8*half + 16*(e/8)
__device__ __forceinline__ v16bf lds_frag_a(const bf16* lA, int m_base, int lane) {
    const int m    = m_base + (lane & 15);
    const int half = lane >> 4;
    const bf16* p  = lA + m * LDP + half * 8;
    FragU r;
    r.u[0] = *(const uint4*)(p);       // k = 8*half + 0..7
    r.u[1] = *(const uint4*)(p + 16);  // k = 8*half + 16..23
    return r.v;
}
// B (32x16 bf16, stored n-major in LDS): lane n = lane%16; k = (e%8) + 8*half + 16*(e/8)
__device__ __forceinline__ v16bf lds_frag_b(const bf16* lB, int n_base, int lane) {
    const int n    = n_base + (lane & 15);
    const int half = lane >> 4;
    const bf16* p  = lB + n * LDP + half * 8;
    FragU r;
    r.u[0] = *(const uint4*)(p);
    r.u[1] = *(const uint4*)(p + 16);
    return r.v;
}

#define WMMA_BF16(acc, a, b) \
    __builtin_amdgcn_wmma_f32_16x16x32_bf16(false, (a), false, (b), (short)0, (acc), false, false)

// =============== generic GEMM: C[M,N] = A[M,K](bf16) * B[N,K](bf16)^T ===============
// 256 thr = 8 waves; block tile 128x128; wave tile 32x64 (2x4 WMMA frags); BK=32.
// Double-buffered LDS; tiles streamed with GLOBAL_LOAD_ASYNC_TO_LDS_B128 when available.
template <int EPI>
__global__ __launch_bounds__(256) void gemm_bf16_wmma(
    const bf16* __restrict__ A, const bf16* __restrict__ Bw,
    const float* __restrict__ Res, void* __restrict__ Out,
    int M, int N, int K)
{
    __shared__ __attribute__((aligned(16))) bf16 lA[2][BM * LDP];
    __shared__ __attribute__((aligned(16))) bf16 lB[2][BNG * LDP];

    const int tid  = threadIdx.x;
    const int lane = tid & 31;
    const int wid  = tid >> 5;
    const int wm   = wid & 3;   // 4 waves over M (32 rows each)
    const int wn   = wid >> 2;  // 2 waves over N (64 cols each)
    const int m0   = blockIdx.y * BM;
    const int n0   = blockIdx.x * BNG;

    // A and B tiles are both 128 rows x 32 k: one row-half (16 bf16) per thread
    const int t_row = tid >> 1, t_seg = (tid & 1) * 16;

    const bf16* gArow = A  + (size_t)(m0 + t_row) * K + t_seg;
    const bf16* gBrow = Bw + (size_t)(n0 + t_row) * K + t_seg;

    v8f acc[2][4];
    v8f zero = {0.f,0.f,0.f,0.f,0.f,0.f,0.f,0.f};
    for (int mi = 0; mi < 2; mi++)
        for (int ni = 0; ni < 4; ni++) acc[mi][ni] = zero;

    const int nk = K / BK;

#if HAVE_ASYNC_LDS
    // prologue: async-fill buffer 0 with tile 0
    async_cp16(gArow,     &lA[0][t_row * LDP + t_seg]);
    async_cp16(gArow + 8, &lA[0][t_row * LDP + t_seg + 8]);
    async_cp16(gBrow,     &lB[0][t_row * LDP + t_seg]);
    async_cp16(gBrow + 8, &lB[0][t_row * LDP + t_seg + 8]);
    for (int kt = 0; kt < nk; kt++) {
        const int cur = kt & 1;
        async_wait0();
        __syncthreads();
        if (kt + 1 < nk) {  // prefetch next tile into the other buffer, under the WMMAs
            const int k1 = (kt + 1) * BK;
            async_cp16(gArow + k1,     &lA[cur ^ 1][t_row * LDP + t_seg]);
            async_cp16(gArow + k1 + 8, &lA[cur ^ 1][t_row * LDP + t_seg + 8]);
            async_cp16(gBrow + k1,     &lB[cur ^ 1][t_row * LDP + t_seg]);
            async_cp16(gBrow + k1 + 8, &lB[cur ^ 1][t_row * LDP + t_seg + 8]);
        }
        v16bf af[2], bfr[4];
        af[0]  = lds_frag_a(lA[cur], wm * 32,      lane);
        af[1]  = lds_frag_a(lA[cur], wm * 32 + 16, lane);
        bfr[0] = lds_frag_b(lB[cur], wn * 64,      lane);
        bfr[1] = lds_frag_b(lB[cur], wn * 64 + 16, lane);
        bfr[2] = lds_frag_b(lB[cur], wn * 64 + 32, lane);
        bfr[3] = lds_frag_b(lB[cur], wn * 64 + 48, lane);
        for (int mi = 0; mi < 2; mi++)
            for (int ni = 0; ni < 4; ni++)
                acc[mi][ni] = WMMA_BF16(acc[mi][ni], af[mi], bfr[ni]);
        __syncthreads();
    }
#else
    for (int kt = 0; kt < nk; kt++) {
        const int cur = kt & 1;
        const int k0  = kt * BK;
        const uint4* gA = (const uint4*)(gArow + k0);
        *(uint4*)(&lA[cur][t_row * LDP + t_seg])     = gA[0];
        *(uint4*)(&lA[cur][t_row * LDP + t_seg + 8]) = gA[1];
        const uint4* gB = (const uint4*)(gBrow + k0);
        *(uint4*)(&lB[cur][t_row * LDP + t_seg])     = gB[0];
        *(uint4*)(&lB[cur][t_row * LDP + t_seg + 8]) = gB[1];
        __syncthreads();
        v16bf af[2], bfr[4];
        af[0]  = lds_frag_a(lA[cur], wm * 32,      lane);
        af[1]  = lds_frag_a(lA[cur], wm * 32 + 16, lane);
        bfr[0] = lds_frag_b(lB[cur], wn * 64,      lane);
        bfr[1] = lds_frag_b(lB[cur], wn * 64 + 16, lane);
        bfr[2] = lds_frag_b(lB[cur], wn * 64 + 32, lane);
        bfr[3] = lds_frag_b(lB[cur], wn * 64 + 48, lane);
        for (int mi = 0; mi < 2; mi++)
            for (int ni = 0; ni < 4; ni++)
                acc[mi][ni] = WMMA_BF16(acc[mi][ni], af[mi], bfr[ni]);
        __syncthreads();
    }
#endif

    const int half = lane >> 4;
    for (int mi = 0; mi < 2; mi++)
        for (int ni = 0; ni < 4; ni++) {
            AccU fa; fa.v = acc[mi][ni];
            const int mb = m0 + wm * 32 + mi * 16 + half * 8;
            const int nn = n0 + wn * 64 + ni * 16 + (lane & 15);
            for (int i = 0; i < 8; i++) {
                const size_t idx = (size_t)(mb + i) * N + nn;
                float vv = fa.f[i];
                if      (EPI == EPI_F32)      ((float*)Out)[idx] = vv;
                else if (EPI == EPI_F32_RES)  ((float*)Out)[idx] = vv + Res[idx];
                else if (EPI == EPI_BF16)     ((bf16*)Out)[idx]  = (bf16)vv;
                else { // erf-based GELU, then bf16
                    float g = 0.5f * vv * (1.0f + erff(vv * 0.70710678118654752f));
                    ((bf16*)Out)[idx] = (bf16)g;
                }
            }
        }
}

// =============== attention scores: S[q,k] = (Q_h · K_h^T)/8, causal-tiled ===============
__global__ __launch_bounds__(256) void attn_scores_kernel(
    const bf16* __restrict__ Q, const bf16* __restrict__ Kb, float* __restrict__ Sc)
{
    const int bh = blockIdx.z, b = bh >> 4, h = bh & 15;
    const int q0 = blockIdx.y * BM;
    const int k0 = blockIdx.x * BNA;
    if (k0 > q0 + (BM - 1)) return;  // tile fully above diagonal (uniform exit)

    __shared__ __attribute__((aligned(16))) bf16 lA[BM * LDP];
    __shared__ __attribute__((aligned(16))) bf16 lB[BNA * LDP];

    const int tid = threadIdx.x, lane = tid & 31, wid = tid >> 5;
    const int wm = wid & 3, wn = wid >> 2;
    const int a_row = tid >> 1, a_seg = (tid & 1) * 16;
    const int b_row = tid >> 2, b_seg = (tid & 3) * 8;

    const bf16* Qb = Q  + (size_t)b * SEQ * D_MODEL + h * D_HEAD;
    const bf16* Kh = Kb + (size_t)b * SEQ * D_MODEL + h * D_HEAD;

    v8f acc[2][2];
    v8f zero = {0.f,0.f,0.f,0.f,0.f,0.f,0.f,0.f};
    acc[0][0]=zero; acc[0][1]=zero; acc[1][0]=zero; acc[1][1]=zero;

    for (int d0 = 0; d0 < D_HEAD; d0 += BK) {
        const uint4* gA = (const uint4*)(Qb + (size_t)(q0 + a_row) * D_MODEL + d0 + a_seg);
        *(uint4*)(lA + a_row * LDP + a_seg)     = gA[0];
        *(uint4*)(lA + a_row * LDP + a_seg + 8) = gA[1];
        const uint4* gB = (const uint4*)(Kh + (size_t)(k0 + b_row) * D_MODEL + d0 + b_seg);
        *(uint4*)(lB + b_row * LDP + b_seg) = gB[0];
        __syncthreads();
        v16bf af0 = lds_frag_a(lA, wm * 32,      lane);
        v16bf af1 = lds_frag_a(lA, wm * 32 + 16, lane);
        v16bf bf0 = lds_frag_b(lB, wn * 32,      lane);
        v16bf bf1 = lds_frag_b(lB, wn * 32 + 16, lane);
        acc[0][0] = WMMA_BF16(acc[0][0], af0, bf0);
        acc[0][1] = WMMA_BF16(acc[0][1], af0, bf1);
        acc[1][0] = WMMA_BF16(acc[1][0], af1, bf0);
        acc[1][1] = WMMA_BF16(acc[1][1], af1, bf1);
        __syncthreads();
    }

    const int half = lane >> 4;
    float* Srow = Sc + (size_t)bh * SEQ * SEQ;
    for (int mi = 0; mi < 2; mi++)
        for (int ni = 0; ni < 2; ni++) {
            AccU fa; fa.v = acc[mi][ni];
            const int qb_ = q0 + wm * 32 + mi * 16 + half * 8;
            const int kk  = k0 + wn * 32 + ni * 16 + (lane & 15);
            for (int i = 0; i < 8; i++)
                Srow[(size_t)(qb_ + i) * SEQ + kk] = 0.125f * fa.f[i];  // 1/sqrt(64)
        }
}

// =============== causal softmax over a row; writes dense bf16 (masked -> 0) ===============
__global__ __launch_bounds__(256) void softmax_kernel(
    const float* __restrict__ Sc, bf16* __restrict__ P)
{
    const int q = blockIdx.x, bh = blockIdx.y, tid = threadIdx.x;
    const float* row = Sc + ((size_t)bh * SEQ + q) * SEQ;
    bf16* prow = P + ((size_t)bh * SEQ + q) * SEQ;

    float v[4]; float mx = -1e30f;
    for (int i = 0; i < 4; i++) {
        int j = tid + i * 256;
        v[i] = (j <= q) ? row[j] : -1e30f;
        mx = fmaxf(mx, v[i]);
    }
    for (int off = 16; off; off >>= 1) mx = fmaxf(mx, __shfl_xor(mx, off, 32));
    __shared__ float sm[8], ss[8];
    if ((tid & 31) == 0) sm[tid >> 5] = mx;
    __syncthreads();
    float m = sm[0];
    for (int i = 1; i < 8; i++) m = fmaxf(m, sm[i]);

    float e[4]; float s = 0.f;
    for (int i = 0; i < 4; i++) {
        int j = tid + i * 256;
        e[i] = (j <= q) ? __expf(v[i] - m) : 0.f;
        s += e[i];
    }
    for (int off = 16; off; off >>= 1) s += __shfl_xor(s, off, 32);
    if ((tid & 31) == 0) ss[tid >> 5] = s;
    __syncthreads();
    float tot = 0.f;
    for (int i = 0; i < 8; i++) tot += ss[i];
    const float inv = 1.0f / tot;
    for (int i = 0; i < 4; i++) {
        int j = tid + i * 256;
        prow[j] = (bf16)(e[i] * inv);
    }
}

// =============== O_h = P(SxS bf16) · V_h(Sx64), causal K-loop bound ===============
__global__ __launch_bounds__(256) void attn_pv_kernel(
    const bf16* __restrict__ P, const bf16* __restrict__ V, bf16* __restrict__ O)
{
    const int bh = blockIdx.z, b = bh >> 4, h = bh & 15;
    const int q0 = blockIdx.y * BM;  // BNA == D_HEAD == 64 -> gridDim.x == 1

    __shared__ __attribute__((aligned(16))) bf16 lA[BM * LDP];
    __shared__ __attribute__((aligned(16))) bf16 lB[D_HEAD * LDP];

    const int tid = threadIdx.x, lane = tid & 31, wid = tid >> 5;
    const int wm = wid & 3, wn = wid >> 2;
    const int a_row = tid >> 1, a_seg = (tid & 1) * 16;
    const int v_kk = tid >> 3, v_d0 = (tid & 7) * 8;   // V staging: transpose to [d][k]

    const bf16* Pb = P + (size_t)bh * SEQ * SEQ;
    const bf16* Vh = V + (size_t)b * SEQ * D_MODEL + h * D_HEAD;

    v8f acc[2][2];
    v8f zero = {0.f,0.f,0.f,0.f,0.f,0.f,0.f,0.f};
    acc[0][0]=zero; acc[0][1]=zero; acc[1][0]=zero; acc[1][1]=zero;

    const int kmax = q0 + BM;  // masked tail is zero-filled by softmax; skip beyond tile diag
    for (int k0 = 0; k0 < kmax; k0 += BK) {
        const uint4* gA = (const uint4*)(Pb + (size_t)(q0 + a_row) * SEQ + k0 + a_seg);
        *(uint4*)(lA + a_row * LDP + a_seg)     = gA[0];
        *(uint4*)(lA + a_row * LDP + a_seg + 8) = gA[1];
        Pack8 tv;
        tv.u = *(const uint4*)(Vh + (size_t)(k0 + v_kk) * D_MODEL + v_d0);
        for (int j = 0; j < 8; j++) lB[(v_d0 + j) * LDP + v_kk] = tv.h[j];
        __syncthreads();
        v16bf af0 = lds_frag_a(lA, wm * 32,      lane);
        v16bf af1 = lds_frag_a(lA, wm * 32 + 16, lane);
        v16bf bf0 = lds_frag_b(lB, wn * 32,      lane);
        v16bf bf1 = lds_frag_b(lB, wn * 32 + 16, lane);
        acc[0][0] = WMMA_BF16(acc[0][0], af0, bf0);
        acc[0][1] = WMMA_BF16(acc[0][1], af0, bf1);
        acc[1][0] = WMMA_BF16(acc[1][0], af1, bf0);
        acc[1][1] = WMMA_BF16(acc[1][1], af1, bf1);
        __syncthreads();
    }

    const int half = lane >> 4;
    for (int mi = 0; mi < 2; mi++)
        for (int ni = 0; ni < 2; ni++) {
            AccU fa; fa.v = acc[mi][ni];
            const int qb_ = q0 + wm * 32 + mi * 16 + half * 8;
            const int dd  = wn * 32 + ni * 16 + (lane & 15);
            for (int i = 0; i < 8; i++)
                O[((size_t)b * SEQ + qb_ + i) * D_MODEL + h * D_HEAD + dd] = (bf16)fa.f[i];
        }
}

// =============== small elementwise kernels ===============
__global__ __launch_bounds__(256) void cvt4_kernel(
    const float* __restrict__ in, bf16* __restrict__ out, size_t n4)
{
    size_t i = (size_t)blockIdx.x * 256 + threadIdx.x;
    if (i >= n4) return;
    float4 f = ((const float4*)in)[i];
    union { bf16 h[4]; uint2 u; } r;
    r.h[0] = (bf16)f.x; r.h[1] = (bf16)f.y; r.h[2] = (bf16)f.z; r.h[3] = (bf16)f.w;
    ((uint2*)out)[i] = r.u;
}

__global__ __launch_bounds__(256) void embed_kernel(
    const int* __restrict__ tok, const float* __restrict__ te,
    const float* __restrict__ pe, float* __restrict__ x)
{
    size_t i = (size_t)blockIdx.x * 256 + threadIdx.x;  // over B*S*D
    int d   = (int)(i & (D_MODEL - 1));
    int row = (int)(i >> 10);        // b*S + s
    int s   = row & (SEQ - 1);
    int t   = tok[row];
    x[i] = te[(size_t)t * D_MODEL + d] + pe[(size_t)s * D_MODEL + d];
}

__global__ __launch_bounds__(256) void rmsnorm_kernel(
    const float* __restrict__ x, const float* __restrict__ w, bf16* __restrict__ out)
{
    const int row = blockIdx.x, tid = threadIdx.x;
    const float* xr = x + (size_t)row * D_MODEL;
    float v[4]; float s = 0.f;
    for (int i = 0; i < 4; i++) { v[i] = xr[tid + i * 256]; s += v[i] * v[i]; }
    for (int off = 16; off; off >>= 1) s += __shfl_xor(s, off, 32);
    __shared__ float sm[8];
    if ((tid & 31) == 0) sm[tid >> 5] = s;
    __syncthreads();
    float tot = 0.f;
    for (int i = 0; i < 8; i++) tot += sm[i];
    const float scale = rsqrtf(RMS_EPS + tot * (1.0f / D_MODEL));
    bf16* orow = out + (size_t)row * D_MODEL;
    for (int i = 0; i < 4; i++) {
        int j = tid + i * 256;
        orow[j] = (bf16)(v[i] * scale * w[j]);
    }
}

// =============== host orchestration ===============
extern "C" void kernel_launch(void* const* d_in, const int* in_sizes, int n_in,
                              void* d_out, int out_size, void* d_ws, size_t ws_size,
                              hipStream_t stream)
{
    const int*   tok  = (const int*)  d_in[0];
    const float* te   = (const float*)d_in[1];
    const float* pe   = (const float*)d_in[2];
    const float* rms1 = (const float*)d_in[3];
    const float* Wq   = (const float*)d_in[4];
    const float* Wk   = (const float*)d_in[5];
    const float* Wv   = (const float*)d_in[6];
    const float* Wo   = (const float*)d_in[7];
    const float* rms2 = (const float*)d_in[8];
    const float* W1   = (const float*)d_in[9];
    const float* W2   = (const float*)d_in[10];
    const float* lnf  = (const float*)d_in[11];
    const float* lmw  = (const float*)d_in[12];

    const size_t DD = (size_t)D_MODEL * D_MODEL;      // 1M
    const size_t DF = (size_t)D_FF * D_MODEL;         // 4M
    const size_t MT = (size_t)BATCH * SEQ;            // 2048 rows
    const size_t BHSS = (size_t)BATCH * N_HEADS * SEQ * SEQ;

    char* bump = (char*)d_ws;
    auto alloc = [&](size_t bytes) -> char* {
        char* r = bump;
        bump += (bytes + 255) & ~(size_t)255;
        return r;
    };
    bf16*  wq_b = (bf16*)alloc(N_LAYERS * DD * 2);
    bf16*  wk_b = (bf16*)alloc(N_LAYERS * DD * 2);
    bf16*  wv_b = (bf16*)alloc(N_LAYERS * DD * 2);
    bf16*  wo_b = (bf16*)alloc(N_LAYERS * DD * 2);
    bf16*  w1_b = (bf16*)alloc(N_LAYERS * DF * 2);
    bf16*  w2_b = (bf16*)alloc(N_LAYERS * DF * 2);
    bf16*  lm_b = (bf16*)alloc((size_t)VOCAB * D_MODEL * 2);
    float* x    = (float*)alloc(MT * D_MODEL * 4);
    bf16*  hb   = (bf16*)alloc(MT * D_MODEL * 2);
    bf16*  qb   = (bf16*)alloc(MT * D_MODEL * 2);
    bf16*  kb   = (bf16*)alloc(MT * D_MODEL * 2);
    bf16*  vb   = (bf16*)alloc(MT * D_MODEL * 2);
    bf16*  ob   = (bf16*)alloc(MT * D_MODEL * 2);
    bf16*  ffb  = (bf16*)alloc(MT * D_FF * 2);
    float* sc   = (float*)alloc(BHSS * 4);
    bf16*  pb   = (bf16*)alloc(BHSS * 2);

    const dim3 blk(256);
    auto cvt = [&](const float* src, bf16* dst, size_t n) {
        size_t n4 = n >> 2;
        cvt4_kernel<<<(unsigned)((n4 + 255) / 256), blk, 0, stream>>>(src, dst, n4);
    };
    cvt(Wq, wq_b, N_LAYERS * DD);
    cvt(Wk, wk_b, N_LAYERS * DD);
    cvt(Wv, wv_b, N_LAYERS * DD);
    cvt(Wo, wo_b, N_LAYERS * DD);
    cvt(W1, w1_b, N_LAYERS * DF);
    cvt(W2, w2_b, N_LAYERS * DF);
    cvt(lmw, lm_b, (size_t)VOCAB * D_MODEL);

    embed_kernel<<<(unsigned)(MT * D_MODEL / 256), blk, 0, stream>>>(tok, te, pe, x);

    const dim3 gSq(D_MODEL / BNG, MT / BM);                           // (8,16)
    const dim3 gFF1(D_FF / BNG, MT / BM);                             // (32,16)
    const dim3 gAttS(SEQ / BNA, SEQ / BM, BATCH * N_HEADS);           // (16,8,32)
    const dim3 gSmax(SEQ, BATCH * N_HEADS);                           // (1024,32)
    const dim3 gPV(1, SEQ / BM, BATCH * N_HEADS);                     // (1,8,32)

    for (int l = 0; l < N_LAYERS; l++) {
        rmsnorm_kernel<<<(unsigned)MT, blk, 0, stream>>>(x, rms1 + (size_t)l * D_MODEL, hb);
        gemm_bf16_wmma<EPI_BF16><<<gSq, blk, 0, stream>>>(hb, wq_b + l * DD, nullptr, qb,
                                                          (int)MT, D_MODEL, D_MODEL);
        gemm_bf16_wmma<EPI_BF16><<<gSq, blk, 0, stream>>>(hb, wk_b + l * DD, nullptr, kb,
                                                          (int)MT, D_MODEL, D_MODEL);
        gemm_bf16_wmma<EPI_BF16><<<gSq, blk, 0, stream>>>(hb, wv_b + l * DD, nullptr, vb,
                                                          (int)MT, D_MODEL, D_MODEL);
        attn_scores_kernel<<<gAttS, blk, 0, stream>>>(qb, kb, sc);
        softmax_kernel<<<gSmax, blk, 0, stream>>>(sc, pb);
        attn_pv_kernel<<<gPV, blk, 0, stream>>>(pb, vb, ob);
        gemm_bf16_wmma<EPI_F32_RES><<<gSq, blk, 0, stream>>>(ob, wo_b + l * DD, x, x,
                                                             (int)MT, D_MODEL, D_MODEL);
        rmsnorm_kernel<<<(unsigned)MT, blk, 0, stream>>>(x, rms2 + (size_t)l * D_MODEL, hb);
        gemm_bf16_wmma<EPI_BF16_GELU><<<gFF1, blk, 0, stream>>>(hb, w1_b + l * DF, nullptr, ffb,
                                                                (int)MT, D_FF, D_MODEL);
        gemm_bf16_wmma<EPI_F32_RES><<<gSq, blk, 0, stream>>>(ffb, w2_b + l * DF, x, x,
                                                             (int)MT, D_MODEL, D_FF);
    }

    rmsnorm_kernel<<<(unsigned)MT, blk, 0, stream>>>(x, lnf, hb);
    const dim3 gHead(VOCAB / BNG, MT / BM);                           // (250,16)
    gemm_bf16_wmma<EPI_F32><<<gHead, blk, 0, stream>>>(hb, lm_b, nullptr, (float*)d_out,
                                                       (int)MT, VOCAB, D_MODEL);
}
